// Tensor_Attention_3977139716817
// MI455X (gfx1250) — matrix operand order repaired and checked by reading, not compiled, not run on hardware
//
#include <hip/hip_runtime.h>
#include <hip/hip_bf16.h>

// ---------------------------------------------------------------------------
// CDNA5 / gfx1250 implementation.  wave32, WMMA 16x16x32 f16 (f32 accum).
// - W and V stored pre-swizzled in B-fragment order: every WMMA B operand is
//   one contiguous 32-byte per-lane read (2 x global_load_b128).
// - GEMM A tiles staged into LDS with GLOBAL_LOAD_ASYNC_TO_LDS_B128 (ASYNCcnt).
// ---------------------------------------------------------------------------

typedef _Float16 half_t;
typedef __attribute__((ext_vector_type(16))) _Float16 v16h;
typedef __attribute__((ext_vector_type(8)))  float    v8f;

#define B_SZ     4
#define L_SZ     2048
#define DM       512
#define NH       8
#define DK       64
#define BH       (B_SZ * NH)        // 32
#define ROWS     (B_SZ * L_SZ)      // 8192
#define LDF      520                // padded row stride (f32 & f16 tiles)

__device__ __forceinline__ v8f wmma_f16(v16h a, v16h b, v8f c) {
    return __builtin_amdgcn_wmma_f32_16x16x32_f16(
        /*neg_a=*/false, a, /*neg_b=*/false, b,
        /*c_mod=*/(short)0, c, /*reuse_a=*/false, /*reuse_b=*/false);
}

// Async copy of 16 bytes / lane from global to LDS.  VDST carries the LDS
// byte address (generic-pointer low 32 bits == LDS offset per flat aperture).
__device__ __forceinline__ void async_copy_b128(unsigned lds_addr, const void* gaddr) {
    asm volatile("global_load_async_to_lds_b128 %0, %1, off"
                 :: "v"(lds_addr), "v"((unsigned long long)gaddr)
                 : "memory");
}
__device__ __forceinline__ void wait_asynccnt0() {
    asm volatile("s_wait_asynccnt 0" ::: "memory");
}

// A fragment: 16x32 f16, row-major source (ld = ldA), base already at k0.
// Contiguous k (not ISA table order): lanes 0-15 hold M=lane, K {0..15}; lanes 16-31 K {16..31}.
__device__ __forceinline__ v16h load_a_frag_h(const half_t* A, int ldA, int lane) {
    int row = lane & 15;
    int kb  = (lane >> 4) << 4;          // 0 or 16
    const half_t* rp = A + (size_t)row * ldA;
    v16h a;
#pragma unroll
    for (int i = 0; i < 8; ++i) {
        int k = kb + 2 * i;   // {kb..kb+15}
        a[2 * i]     = rp[k];
        a[2 * i + 1] = rp[k + 1];
    }
    return a;
}

// Same A fragment but converting from f32 source on the fly.
__device__ __forceinline__ v16h load_a_frag_f32(const float* A, int ldA, int lane) {
    int row = lane & 15;
    int kb  = (lane >> 4) << 4;
    const float* rp = A + (size_t)row * ldA;
    v16h a;
#pragma unroll
    for (int i = 0; i < 8; ++i) {
        int k = kb + 2 * i;
        a[2 * i]     = (half_t)rp[k];
        a[2 * i + 1] = (half_t)rp[k + 1];
    }
    return a;
}

// B fragment from swizzled storage: layout [tile][lane][16], b[t] <-> K=kb+t,
// N=lane&15 with kb=16*(lane>=16).  One contiguous 32-byte per-lane read.
__device__ __forceinline__ v16h load_b_frag_sw(const half_t* tile_base, int lane) {
    const half_t* p = tile_base + (size_t)lane * 16;
    v16h b;
#pragma unroll
    for (int i = 0; i < 16; ++i) b[i] = p[i];
    return b;
}

// B fragment where logical B = Xt^T, Xt is N x K row-major (e.g. K-matrix
// [keys, d]): B[k][n] = Xt[n][k].  16 contiguous f16 per lane.
__device__ __forceinline__ v16h load_bT_frag(const half_t* Xt, int ldX, int lane) {
    int col = lane & 15;
    int kb  = (lane >> 4) << 4;
    const half_t* rp = Xt + (size_t)col * ldX + kb;
    v16h b;
#pragma unroll
    for (int i = 0; i < 16; ++i) b[i] = rp[i];
    return b;
}

// Swizzled W tile offset: 32 n-tiles (jt) x 16 k-chunks (kc), 32 lanes x 16 halfs.
__device__ __forceinline__ size_t w_tile_off(int jt, int kc) {
    return (((size_t)jt * 16 + kc) * 32) * 16;
}

// Stage a 16 x 512 f32 row-block of X into LDS (padded stride LDF) with the
// async data mover, then convert once to f16 (padded stride LDF).
__device__ __forceinline__ void stage_a_tile(const float* __restrict__ X, int r0,
                                             float* Af32, half_t* As, int tid) {
    unsigned lds_base = (unsigned)(unsigned long long)(const void*)Af32;
#pragma unroll
    for (int it = 0; it < 8; ++it) {
        int idx = it * 256 + tid;          // 2048 chunks of 4 floats
        int r = idx >> 7, c4 = idx & 127;
        unsigned laddr = lds_base + (unsigned)(r * LDF + c4 * 4) * 4u;
        const void* g = (const void*)(X + (size_t)(r0 + r) * DM + c4 * 4);
        async_copy_b128(laddr, g);
    }
    wait_asynccnt0();
    __syncthreads();
#pragma unroll
    for (int it = 0; it < 8; ++it) {
        int idx = it * 256 + tid;
        int r = idx >> 7, c4 = idx & 127;
        const float* s = &Af32[r * LDF + c4 * 4];
        half_t* d = &As[r * LDF + c4 * 4];
        d[0] = (half_t)s[0]; d[1] = (half_t)s[1];
        d[2] = (half_t)s[2]; d[3] = (half_t)s[3];
    }
    __syncthreads();
}

// ---------------------------------------------------------------------------
// Kernel 1: collapse TT cores into dense 512x512 W, stored SWIZZLED (f16).
// W[i][j] = sum_{u,v,r} g1[w,o,u] g2[u,x,p,v] g3[v,y,q,r] g4[r,z,s]
//   i = w*128 + x*32 + y*4 + z (K dim) ; j = o*128 + p*32 + q*4 + s (N dim)
// ---------------------------------------------------------------------------
struct TTCores { const float *g1, *g2, *g3, *g4; };
struct TTAll   { TTCores c[4]; };

__global__ __launch_bounds__(256) void build_w_kernel(TTAll all, half_t* __restrict__ Wh) {
    __shared__ float  A1[5120];     // (((w*4+o)*4+x)*4+p)*20 + v
    __shared__ half_t B2[20480];    // v*1024 + ((y*8+q)*4+z)*4 + s
    const TTCores cc = all.c[blockIdx.x];
    const int tid = threadIdx.x;

    for (int idx = tid; idx < 5120; idx += 256) {
        int v = idx % 20, t = idx / 20;
        int p = t & 3, x = (t >> 2) & 3, o = (t >> 4) & 3, w = (t >> 6) & 3;
        float s = 0.f;
        for (int u = 0; u < 20; ++u)
            s += cc.g1[(w * 4 + o) * 20 + u] * cc.g2[((u * 4 + x) * 4 + p) * 20 + v];
        A1[idx] = s;
    }
    for (int idx = tid; idx < 20480; idx += 256) {
        int v = idx >> 10, rem = idx & 1023;
        int s4 = rem & 3, z = (rem >> 2) & 3, q = (rem >> 4) & 7, y = (rem >> 7) & 7;
        float s = 0.f;
        for (int r = 0; r < 20; ++r)
            s += cc.g3[((v * 8 + y) * 8 + q) * 20 + r] * cc.g4[(r * 4 + z) * 4 + s4];
        B2[idx] = (half_t)s;
    }
    __syncthreads();

    half_t* W = Wh + (size_t)blockIdx.x * DM * DM;
    for (int e = tid; e < DM * DM; e += 256) {
        int i = e >> 9, j = e & 511;                 // i = K index, j = N index
        int w = i >> 7, x = (i >> 5) & 3, y = (i >> 2) & 7, z = i & 3;
        int o = j >> 7, p = (j >> 5) & 3, q = (j >> 2) & 7, s4 = j & 3;
        const float*  a1 = &A1[(((w * 4 + o) * 4 + x) * 4 + p) * 20];
        const half_t* b2 = &B2[((y * 8 + q) * 4 + z) * 4 + s4];
        float acc = 0.f;
#pragma unroll 4
        for (int v = 0; v < 20; ++v) acc += a1[v] * (float)b2[v * 1024];

        // swizzled B-fragment store: t = i&15, lane = 16*((i>>4)&1) + (j&15)
        int kc = i >> 5, tt = i & 15, lnhi = (i >> 4) & 1;
        int jt = j >> 4, cl = j & 15;
        W[w_tile_off(jt, kc) + (size_t)(lnhi * 16 + cl) * 16 + tt] = (half_t)acc;
    }
}

// ---------------------------------------------------------------------------
// Kernel 2: projection GEMM.  X[8192,512] f32 @ Wsw + bias.
// Block = 256 thr (8 waves); each wave computes TWO 16x16 tiles (jt0, jt0+1).
// grid (ROWS/16, 2).  vswizzle=0: out [B,H,L,64] f16 row-major (Q/K).
// vswizzle=1: out per-head B-fragment swizzled [kc][nt][lane][16] (V).
// ---------------------------------------------------------------------------
__global__ __launch_bounds__(256) void gemm_proj_kernel(const float* __restrict__ X,
                                                        const half_t* __restrict__ Wsw,
                                                        const float* __restrict__ bias,
                                                        half_t* __restrict__ Out,
                                                        int vswizzle) {
    __shared__ float  Af32[16 * LDF];
    __shared__ half_t As[16 * LDF];
    const int tid = threadIdx.x, lane = tid & 31, wv = tid >> 5;
    const int r0 = blockIdx.x * 16;

    stage_a_tile(X, r0, Af32, As, tid);

    const int jt0 = blockIdx.y * 16 + wv * 2;      // two adjacent n-tiles
    v8f acc0 = {0.f, 0.f, 0.f, 0.f, 0.f, 0.f, 0.f, 0.f};
    v8f acc1 = {0.f, 0.f, 0.f, 0.f, 0.f, 0.f, 0.f, 0.f};
    for (int kc = 0; kc < 16; ++kc) {
        if (kc + 1 < 16)
            __builtin_prefetch(Wsw + w_tile_off(jt0, kc + 1), 0, 1);   // global_prefetch_b8
        v16h a  = load_a_frag_h(&As[kc * 32], LDF, lane);
        v16h b0 = load_b_frag_sw(Wsw + w_tile_off(jt0,     kc), lane);
        v16h b1 = load_b_frag_sw(Wsw + w_tile_off(jt0 + 1, kc), lane);
        acc0 = wmma_f16(a, b0, acc0);
        acc1 = wmma_f16(a, b1, acc1);
    }

    const int rowbase = (lane >> 4) << 3;
#pragma unroll
    for (int t = 0; t < 2; ++t) {
        v8f acc = t ? acc1 : acc0;
        int col = (jt0 + t) * 16 + (lane & 15);
        float bc = bias[col];
        int h = col >> 6, d = col & 63;
#pragma unroll
        for (int r = 0; r < 8; ++r) {
            int grow = r0 + rowbase + r;
            int bb = grow >> 11, l = grow & 2047;
            size_t head = ((size_t)bb * NH + h) * (L_SZ * DK);
            float val = acc[r] + bc;
            if (vswizzle) {
                // V as B-fragment: K = l, N = d within head.
                int kc = l >> 5, tt = l & 15, lnhi = (l >> 4) & 1;
                int nt = d >> 4, cl = d & 15;
                Out[head + (((size_t)kc * 4 + nt) * 32 + lnhi * 16 + cl) * 16 + tt] =
                    (half_t)val;
            } else {
                Out[head + (size_t)l * DK + d] = (half_t)val;
            }
        }
    }
}

// ---------------------------------------------------------------------------
// Kernel 3: S = (Q K^T) / sqrt(64) -> raw scores into attn buffer (f32).
// grid (L/16, L/256, B*H), block 256; each wave does TWO 16x16 tiles.
// ---------------------------------------------------------------------------
__global__ __launch_bounds__(256) void attn_qk_kernel(const half_t* __restrict__ Qp,
                                                      const half_t* __restrict__ Kp,
                                                      float* __restrict__ attn) {
    const int tid = threadIdx.x, lane = tid & 31, wv = tid >> 5;
    const int bh = blockIdx.z;
    const half_t* Q = Qp + (size_t)bh * L_SZ * DK;
    const half_t* K = Kp + (size_t)bh * L_SZ * DK;
    const int m0 = blockIdx.x * 16;
    const int n0 = blockIdx.y * 256 + wv * 32;      // two adjacent key tiles

    v16h a0  = load_a_frag_h(Q + (size_t)m0 * DK,            DK, lane);
    v16h a1  = load_a_frag_h(Q + (size_t)m0 * DK + 32,       DK, lane);
    v16h b00 = load_bT_frag(K + (size_t)n0 * DK,             DK, lane);
    v16h b01 = load_bT_frag(K + (size_t)n0 * DK + 32,        DK, lane);
    v16h b10 = load_bT_frag(K + (size_t)(n0 + 16) * DK,      DK, lane);
    v16h b11 = load_bT_frag(K + (size_t)(n0 + 16) * DK + 32, DK, lane);

    v8f acc0 = {0.f, 0.f, 0.f, 0.f, 0.f, 0.f, 0.f, 0.f};
    v8f acc1 = {0.f, 0.f, 0.f, 0.f, 0.f, 0.f, 0.f, 0.f};
    acc0 = wmma_f16(a0, b00, acc0);
    acc0 = wmma_f16(a1, b01, acc0);
    acc1 = wmma_f16(a0, b10, acc1);
    acc1 = wmma_f16(a1, b11, acc1);

    float* S = attn + (size_t)bh * L_SZ * L_SZ;
    const int col = lane & 15, rowbase = (lane >> 4) << 3;
#pragma unroll
    for (int r = 0; r < 8; ++r) {
        size_t rowoff = (size_t)(m0 + rowbase + r) * L_SZ;
        S[rowoff + n0 + col]      = acc0[r] * 0.125f;
        S[rowoff + n0 + 16 + col] = acc1[r] * 0.125f;
    }
}

// ---------------------------------------------------------------------------
// Kernel 4: in-place row softmax over 2048 scores.  1 block / row.
// ---------------------------------------------------------------------------
__global__ __launch_bounds__(256) void attn_softmax_kernel(float* __restrict__ attn) {
    __shared__ float red[256];
    const int tid = threadIdx.x;
    float* p = attn + (size_t)blockIdx.x * L_SZ;

    float v[8];
    float mx = -3.4e38f;
#pragma unroll
    for (int i = 0; i < 8; ++i) { v[i] = p[tid * 8 + i]; mx = fmaxf(mx, v[i]); }
    red[tid] = mx; __syncthreads();
    for (int off = 128; off > 0; off >>= 1) {
        if (tid < off) red[tid] = fmaxf(red[tid], red[tid + off]);
        __syncthreads();
    }
    mx = red[0]; __syncthreads();

    float s = 0.f;
#pragma unroll
    for (int i = 0; i < 8; ++i) { v[i] = __expf(v[i] - mx); s += v[i]; }
    red[tid] = s; __syncthreads();
    for (int off = 128; off > 0; off >>= 1) {
        if (tid < off) red[tid] += red[tid + off];
        __syncthreads();
    }
    const float inv = 1.f / red[0];
#pragma unroll
    for (int i = 0; i < 8; ++i) p[tid * 8 + i] = v[i] * inv;
}

// ---------------------------------------------------------------------------
// Kernel 5: O = P @ V.  grid (L/16, 1, B*H), block 128 (4 waves x 16 cols).
// V is stored swizzled: [kc(64)][nt(4)][lane(32)][16].
// ---------------------------------------------------------------------------
__global__ __launch_bounds__(128) void attn_pv_kernel(const float* __restrict__ attn,
                                                      const half_t* __restrict__ Vsw,
                                                      float* __restrict__ O) {
    const int tid = threadIdx.x, lane = tid & 31, wv = tid >> 5;
    const int bh = blockIdx.z;
    const int m0 = blockIdx.x * 16;
    const float*  P = attn + (size_t)bh * L_SZ * L_SZ + (size_t)m0 * L_SZ;
    const half_t* V = Vsw + (size_t)bh * L_SZ * DK;   // swizzled within head

    v8f acc = {0.f, 0.f, 0.f, 0.f, 0.f, 0.f, 0.f, 0.f};
    for (int kc = 0; kc < 64; ++kc) {
        v16h a = load_a_frag_f32(P + kc * 32, L_SZ, lane);
        v16h b = load_b_frag_sw(V + (((size_t)kc * 4 + wv) * 32) * 16, lane);
        acc = wmma_f16(a, b, acc);
    }

    const int bb = bh >> 3, h = bh & 7;
    const int col = h * DK + wv * 16 + (lane & 15);
    const int rowbase = (lane >> 4) << 3;
#pragma unroll
    for (int r = 0; r < 8; ++r)
        O[(size_t)(bb * L_SZ + m0 + rowbase + r) * DM + col] = acc[r];
}

// ---------------------------------------------------------------------------
// Kernel 6: fc GEMM + bias + residual -> pre-LN (f32 [8192,512]).
// ---------------------------------------------------------------------------
__global__ __launch_bounds__(256) void gemm_fc_kernel(const float* __restrict__ X,
                                                      const half_t* __restrict__ Wsw,
                                                      const float* __restrict__ bias,
                                                      const float* __restrict__ resid,
                                                      float* __restrict__ OutPre) {
    __shared__ float  Af32[16 * LDF];
    __shared__ half_t As[16 * LDF];
    const int tid = threadIdx.x, lane = tid & 31, wv = tid >> 5;
    const int r0 = blockIdx.x * 16;

    stage_a_tile(X, r0, Af32, As, tid);

    const int jt0 = blockIdx.y * 16 + wv * 2;
    v8f acc0 = {0.f, 0.f, 0.f, 0.f, 0.f, 0.f, 0.f, 0.f};
    v8f acc1 = {0.f, 0.f, 0.f, 0.f, 0.f, 0.f, 0.f, 0.f};
    for (int kc = 0; kc < 16; ++kc) {
        if (kc + 1 < 16)
            __builtin_prefetch(Wsw + w_tile_off(jt0, kc + 1), 0, 1);
        v16h a  = load_a_frag_h(&As[kc * 32], LDF, lane);
        v16h b0 = load_b_frag_sw(Wsw + w_tile_off(jt0,     kc), lane);
        v16h b1 = load_b_frag_sw(Wsw + w_tile_off(jt0 + 1, kc), lane);
        acc0 = wmma_f16(a, b0, acc0);
        acc1 = wmma_f16(a, b1, acc1);
    }

    const int rowbase = (lane >> 4) << 3;
#pragma unroll
    for (int t = 0; t < 2; ++t) {
        v8f acc = t ? acc1 : acc0;
        int col = (jt0 + t) * 16 + (lane & 15);
        float bc = bias[col];
#pragma unroll
        for (int r = 0; r < 8; ++r) {
            size_t idx = (size_t)(r0 + rowbase + r) * DM + col;
            OutPre[idx] = acc[r] + bc + resid[idx];
        }
    }
}

// ---------------------------------------------------------------------------
// Kernel 7: layernorm over 512, 1 block / row (2 elems / thread).
// ---------------------------------------------------------------------------
__global__ __launch_bounds__(256) void ln_kernel(const float* __restrict__ X,
                                                 const float* __restrict__ g,
                                                 const float* __restrict__ b,
                                                 float* __restrict__ out) {
    __shared__ float red[256];
    const int tid = threadIdx.x;
    const size_t base = (size_t)blockIdx.x * DM;
    float x0 = X[base + tid], x1 = X[base + 256 + tid];

    red[tid] = x0 + x1; __syncthreads();
    for (int off = 128; off > 0; off >>= 1) {
        if (tid < off) red[tid] += red[tid + off];
        __syncthreads();
    }
    const float mu = red[0] * (1.f / DM); __syncthreads();

    float d0 = x0 - mu, d1 = x1 - mu;
    red[tid] = d0 * d0 + d1 * d1; __syncthreads();
    for (int off = 128; off > 0; off >>= 1) {
        if (tid < off) red[tid] += red[tid + off];
        __syncthreads();
    }
    const float inv = rsqrtf(red[0] * (1.f / DM) + 1e-12f);
    out[base + tid]       = d0 * inv * g[tid]       + b[tid];
    out[base + 256 + tid] = d1 * inv * g[tid + 256] + b[tid + 256];
}

// ---------------------------------------------------------------------------
// Host-side launch.
// ---------------------------------------------------------------------------
extern "C" void kernel_launch(void* const* d_in, const int* in_sizes, int n_in,
                              void* d_out, int out_size, void* d_ws, size_t ws_size,
                              hipStream_t stream) {
    (void)in_sizes; (void)n_in; (void)out_size; (void)ws_size;

    // Inputs: [wq|wk|wv|fc] x {g1,g2,g3,g4,b}, then q,k,v,ln_g,ln_b.
    TTAll tta;
    const float* biases[4];
    for (int w = 0; w < 4; ++w) {
        tta.c[w].g1 = (const float*)d_in[w * 5 + 0];
        tta.c[w].g2 = (const float*)d_in[w * 5 + 1];
        tta.c[w].g3 = (const float*)d_in[w * 5 + 2];
        tta.c[w].g4 = (const float*)d_in[w * 5 + 3];
        biases[w]   = (const float*)d_in[w * 5 + 4];
    }
    const float* q    = (const float*)d_in[20];
    const float* k    = (const float*)d_in[21];
    const float* v    = (const float*)d_in[22];
    const float* ln_g = (const float*)d_in[23];
    const float* ln_b = (const float*)d_in[24];

    float* outp = (float*)d_out;                                   // [B,L,512]
    float* attn = (float*)d_out + (size_t)B_SZ * L_SZ * DM;        // [B,H,L,L]

    // Workspace layout.
    char* ws = (char*)d_ws;
    size_t off = 0;
    half_t* Wh = (half_t*)(ws + off); off += (size_t)4 * DM * DM * sizeof(half_t);   // 2 MB
    half_t* Qp = (half_t*)(ws + off); off += (size_t)BH * L_SZ * DK * sizeof(half_t);
    half_t* Kp = (half_t*)(ws + off); off += (size_t)BH * L_SZ * DK * sizeof(half_t);
    half_t* Vp = (half_t*)(ws + off); off += (size_t)BH * L_SZ * DK * sizeof(half_t);
    float*  O  = (float*)(ws + off);  off += (size_t)ROWS * DM * sizeof(float);
    float*  pre= (float*)(ws + off);  off += (size_t)ROWS * DM * sizeof(float);

    // 1. Materialize dense TT weights (f16, swizzled fragment layout).
    build_w_kernel<<<4, 256, 0, stream>>>(tta, Wh);

    // 2. Q/K/V projections (WMMA GEMM).  V output is B-fragment swizzled.
    dim3 ggrid(ROWS / 16, 2);
    gemm_proj_kernel<<<ggrid, 256, 0, stream>>>(q, Wh + 0 * (size_t)DM * DM, biases[0], Qp, 0);
    gemm_proj_kernel<<<ggrid, 256, 0, stream>>>(k, Wh + 1 * (size_t)DM * DM, biases[1], Kp, 0);
    gemm_proj_kernel<<<ggrid, 256, 0, stream>>>(v, Wh + 2 * (size_t)DM * DM, biases[2], Vp, 1);

    // 3. Attention scores -> attn output buffer.
    attn_qk_kernel<<<dim3(L_SZ / 16, L_SZ / 256, BH), 256, 0, stream>>>(Qp, Kp, attn);

    // 4. Row softmax (in place, normalized attn is a reference output).
    attn_softmax_kernel<<<dim3(BH * L_SZ), 256, 0, stream>>>(attn);

    // 5. O = P @ V (WMMA, swizzled V).
    attn_pv_kernel<<<dim3(L_SZ / 16, 1, BH), 128, 0, stream>>>(attn, Vp, O);

    // 6. Output projection + residual.
    gemm_fc_kernel<<<ggrid, 256, 0, stream>>>(O, Wh + 3 * (size_t)DM * DM, biases[3], q, pre);

    // 7. LayerNorm -> final output.
    ln_kernel<<<8192, 256, 0, stream>>>(pre, ln_g, ln_b, outp);
}
